// TaylorKANLayer_47218870453028
// MI455X (gfx1250) — matrix operand order repaired and checked by reading, not compiled, not run on hardware
//
#include <hip/hip_runtime.h>
#include <stdint.h>

#ifndef __has_builtin
#define __has_builtin(x) 0
#endif

#define GAS __attribute__((address_space(1)))
#define LAS __attribute__((address_space(3)))

#if __has_builtin(__builtin_amdgcn_global_load_async_to_lds_b128)
#define HAVE_ASYNC_LDS 1
#else
#define HAVE_ASYNC_LDS 0
#endif

// ---------------------------------------------------------------------------
// TaylorKAN layer == GEMM:  out[b,o] = sum_{dk} P[b,dk] * W[o,dk] + bias[o]
//   P[b, d*4+k] = x[b,d]^k   (expanded on the fly into bf16 LDS tiles)
//   W[o, d*4+k] = coeffs[o,d,k]  (row-major over [o][dk] already; B operand of
//                                 WMMA is KxN col-major == NxK row-major == W)
// M=8192, N=1024, K=4096.  v_wmma_f32_16x16x32_bf16, f32 accumulate.
// B tile: global_load_async_to_lds_b128 DMA, double-buffered LDS.
// ---------------------------------------------------------------------------

typedef __attribute__((ext_vector_type(16))) __bf16 v16bf;
typedef __attribute__((ext_vector_type(8)))  float  v8f;
typedef __attribute__((ext_vector_type(4)))  int    v4i;

#define BM   128
#define BN   128
#define BK   64
#define LDA  72      // BK + 8 halves pad -> 36-dword row stride, bank-conflict free
#define LDB  72
#define DIN  1024
#define KTOT 4096
#define NOUT 1024
#define MROWS 8192
#define NSTAGES (KTOT / BK)   // 64

static __device__ __forceinline__ uint32_t pack2bf(float a, float b) {
#if defined(__gfx1250__)
  // Backend provably supports this mnemonic (it emits it itself, but only in
  // single-source form). Two-source form packs a->[15:0], b->[31:16] in one op.
  uint32_t r;
  asm("v_cvt_pk_bf16_f32 %0, %1, %2" : "=v"(r) : "v"(a), "v"(b));
  return r;
#else
  union { __bf16 h[2]; uint32_t u; } p;
  p.h[0] = (__bf16)a;
  p.h[1] = (__bf16)b;
  return p.u;
#endif
}

union FragU { uint4 q[2]; v16bf v; };

#if HAVE_ASYNC_LDS
static __device__ __forceinline__ void wait_async_zero() {
#if __has_builtin(__builtin_amdgcn_s_wait_asynccnt)
  __builtin_amdgcn_s_wait_asynccnt(0);
#else
  asm volatile("s_wait_asynccnt 0" ::: "memory");
#endif
}
#endif

// One-time f32 -> bf16 conversion of coeffs into workspace (8 MB).
__global__ void __launch_bounds__(256) wb_convert(const float* __restrict__ src,
                                                  unsigned short* __restrict__ dst,
                                                  int n4) {
  int i = blockIdx.x * 256 + threadIdx.x;
  if (i >= n4) return;
  const float4 f = ((const float4*)src)[i];
  uint2 o;
  o.x = pack2bf(f.x, f.y);
  o.y = pack2bf(f.z, f.w);
  ((uint2*)dst)[i] = o;
}

template <bool PRECONV>
__global__ void __launch_bounds__(256, 1) taylor_gemm(
    const float* __restrict__ x,       // [8192][1024] f32
    const void*  __restrict__ wsrc,    // PRECONV ? bf16 W[1024][4096] : f32 coeffs
    const float* __restrict__ bias,    // [1024]
    float*       __restrict__ out)     // [8192][1024]
{
  __shared__ __align__(16) unsigned short As[2][BM * LDA];   // 2 x 18 KB
  __shared__ __align__(16) unsigned short Bs[2][BN * LDB];   // 2 x 18 KB

  const int tid  = threadIdx.x;
  const int lane = tid & 31;
  const int wid  = tid >> 5;          // 8 waves
  const int wm   = wid & 1;           // 2 waves along M
  const int wn   = wid >> 1;          // 4 waves along N
  const int m0   = blockIdx.x * BM;
  const int n0   = blockIdx.y * BN;

  // ---- A fill: 128 rows x 16 d per stage; thread -> (row, d-pair), 4 iters ----
  const int a_row = tid >> 3;         // 0..31  (+32*i)
  const int a_dp  = tid & 7;          // d-pair within the 16-d group
  const float* xp = x + (size_t)(m0 + a_row) * DIN + a_dp * 2;

  // ---- B fill: 128 rows x 64 halves per stage; 2 threads/row, 32 halves each ----
  const int b_row = tid >> 1;
  const int b_seg = tid & 1;
  const unsigned short* wb_row =
      (const unsigned short*)wsrc + (size_t)(n0 + b_row) * KTOT + b_seg * 32;
  const float* wf_row =
      (const float*)wsrc + (size_t)(n0 + b_row) * KTOT + b_seg * 32;

  float2 xr[4];
  uint4  br[4];

  // Issue global work for stage s1 (async B goes straight to LDS buffer q).
  auto fill_issue = [&](int q, int s1) {
#pragma unroll
    for (int i = 0; i < 4; ++i)
      xr[i] = *(const float2*)(xp + (size_t)i * 32 * DIN + s1 * (BK / 4));
#if HAVE_ASYNC_LDS
    if constexpr (PRECONV) {
      GAS v4i* src = (GAS v4i*)(wb_row + s1 * BK);
      LAS v4i* dst = (LAS v4i*)&Bs[q][b_row * LDB + b_seg * 32];
      __builtin_amdgcn_global_load_async_to_lds_b128(src, dst, 0, 0);
      __builtin_amdgcn_global_load_async_to_lds_b128(src, dst, 16, 0);
      __builtin_amdgcn_global_load_async_to_lds_b128(src, dst, 32, 0);
      __builtin_amdgcn_global_load_async_to_lds_b128(src, dst, 48, 0);
      return;
    }
#endif
    if (PRECONV) {
      const unsigned short* p = wb_row + s1 * BK;
#pragma unroll
      for (int j = 0; j < 4; ++j) br[j] = ((const uint4*)p)[j];
    } else {
      const float* p = wf_row + s1 * BK;
#pragma unroll
      for (int j = 0; j < 4; ++j) {
        float4 f0 = ((const float4*)p)[2 * j];
        float4 f1 = ((const float4*)p)[2 * j + 1];
        br[j].x = pack2bf(f0.x, f0.y);
        br[j].y = pack2bf(f0.z, f0.w);
        br[j].z = pack2bf(f1.x, f1.y);
        br[j].w = pack2bf(f1.z, f1.w);
      }
    }
  };

  // Commit staged registers to LDS buffer q (async B already landed in LDS).
  auto fill_commit = [&](int q) {
#pragma unroll
    for (int i = 0; i < 4; ++i) {
      const float ax = xr[i].x, ay = xr[i].y;
      uint4 o;
      o.x = pack2bf(1.0f, ax);
      o.y = pack2bf(ax * ax, ax * ax * ax);
      o.z = pack2bf(1.0f, ay);
      o.w = pack2bf(ay * ay, ay * ay * ay);
      *(uint4*)&As[q][(a_row + i * 32) * LDA + a_dp * 8] = o;
    }
#if HAVE_ASYNC_LDS
    if constexpr (PRECONV) {
      wait_async_zero();   // B tile DMA complete before the barrier
      return;
    }
#endif
#pragma unroll
    for (int j = 0; j < 4; ++j)
      *(uint4*)&Bs[q][b_row * LDB + b_seg * 32 + j * 8] = br[j];
  };

  v8f acc[4][2];
#pragma unroll
  for (int mi = 0; mi < 4; ++mi)
#pragma unroll
    for (int ni = 0; ni < 2; ++ni)
      acc[mi][ni] = (v8f){0.f, 0.f, 0.f, 0.f, 0.f, 0.f, 0.f, 0.f};

  const int r16 = lane & 15;
  const int hh  = lane >> 4;

  auto compute_stage = [&](int p) {
#pragma unroll
    for (int kk = 0; kk < 2; ++kk) {
      const int kb = kk * 32;
      FragU a[4], b[2];
#pragma unroll
      for (int mi = 0; mi < 4; ++mi) {
        // A 16x32 bf16 frag: lane<16 holds K 0..7 & 16..23; lane>=16: 8..15 & 24..31
        const unsigned short* pa =
            &As[p][(wm * 64 + mi * 16 + r16) * LDA + kb + hh * 8];
        a[mi].q[0] = *(const uint4*)pa;
        a[mi].q[1] = *(const uint4*)(pa + 16);
      }
#pragma unroll
      for (int ni = 0; ni < 2; ++ni) {
        // B 32x16 bf16 frag via N-major rows: lane<16 K 0..7 & 16..23; lane>=16: 8..15 & 24..31
        const unsigned short* pb =
            &Bs[p][(wn * 32 + ni * 16 + r16) * LDB + kb + hh * 8];
        b[ni].q[0] = *(const uint4*)pb;
        b[ni].q[1] = *(const uint4*)(pb + 16);
      }
#pragma unroll
      for (int mi = 0; mi < 4; ++mi)
#pragma unroll
        for (int ni = 0; ni < 2; ++ni)
          acc[mi][ni] = __builtin_amdgcn_wmma_f32_16x16x32_bf16(
              false, a[mi].v, false, b[ni].v, (short)0, acc[mi][ni], false, false);
    }
  };

  // ---- prologue: fill buffer 0 ----
  fill_issue(0, 0);
  fill_commit(0);
  __syncthreads();

  // ---- pipelined main loop: async/next-stage loads overlap current WMMAs ----
#pragma unroll 2
  for (int s = 0; s < NSTAGES; ++s) {
    const int p = s & 1;
    const bool have_next = (s + 1 < NSTAGES);
    if (have_next) fill_issue(p ^ 1, s + 1);   // DMA into other buffer, x -> regs
    compute_stage(p);
    if (have_next) fill_commit(p ^ 1);
    __syncthreads();
  }

  // ---- epilogue: D layout 16x16 f32 -> row = v + 8*(lane>=16), col = lane&15 ----
#pragma unroll
  for (int mi = 0; mi < 4; ++mi) {
#pragma unroll
    for (int ni = 0; ni < 2; ++ni) {
      const int gm = m0 + wm * 64 + mi * 16 + hh * 8;
      const int gn = n0 + wn * 32 + ni * 16 + r16;
      const float bv = bias[gn];
      float* po = &out[(size_t)gm * NOUT + gn];
#pragma unroll
      for (int v = 0; v < 8; ++v)
        po[(size_t)v * NOUT] = acc[mi][ni][v] + bv;
    }
  }
}

extern "C" void kernel_launch(void* const* d_in, const int* in_sizes, int n_in,
                              void* d_out, int out_size, void* d_ws, size_t ws_size,
                              hipStream_t stream) {
  const float* x      = (const float*)d_in[0];   // 8192*1024
  const float* coeffs = (const float*)d_in[1];   // 1024*1024*4
  const float* bias   = (const float*)d_in[2];   // 1024
  float* out = (float*)d_out;

  const size_t wb_bytes = (size_t)NOUT * KTOT * sizeof(unsigned short);  // 8 MB
  dim3 grid(MROWS / BM, NOUT / BN);   // 64 x 8

  if (ws_size >= wb_bytes) {
    unsigned short* Wb = (unsigned short*)d_ws;
    const int n4 = NOUT * KTOT / 4;
    wb_convert<<<(n4 + 255) / 256, 256, 0, stream>>>(coeffs, Wb, n4);
    taylor_gemm<true><<<grid, 256, 0, stream>>>(x, Wb, bias, out);
  } else {
    taylor_gemm<false><<<grid, 256, 0, stream>>>(x, coeffs, bias, out);
  }
}
